// WIPESImage_RS_70506183131599
// MI455X (gfx1250) — matrix operand order repaired and checked by reading, not compiled, not run on hardware
//
#include <hip/hip_runtime.h>
#include <hip/hip_bf16.h>

typedef __attribute__((ext_vector_type(16))) _Float16 v16h;
typedef __attribute__((ext_vector_type(8)))  float    v8f;

#define IMG_W 512
#define IMG_H 512
#define TILE_W 16
#define TILE_H 8
#define NGAUSS 4096
#define CAP 320                    // LDS survivor buffer: 256 new + <32 carry
#define ALPHA_THRESH (1.0f / 255.0f)

// ---------------------------------------------------------------------------
// Preprocess: per-gaussian projection -> SoA float4 arrays in workspace.
//   gA = (cx, cy, conA, conB)
//   gB = (conC, fx, fy, opacity)
//   gC = (r, g, b, cullRadius)
// ---------------------------------------------------------------------------
__global__ __launch_bounds__(256) void wipes_preprocess(
    const float* __restrict__ xyz_raw, const float* __restrict__ scaling_raw,
    const float* __restrict__ rotation_raw, const float* __restrict__ normf_raw,
    const float* __restrict__ features, const float* __restrict__ opacity,
    float4* __restrict__ gA, float4* __restrict__ gB, float4* __restrict__ gC,
    int n) {
  int g = blockIdx.x * blockDim.x + threadIdx.x;
  if (g >= n) return;

  float tx = tanhf(xyz_raw[2 * g + 0]);
  float ty = tanhf(xyz_raw[2 * g + 1]);
  float cx = 0.5f * ((tx + 1.0f) * (float)IMG_W - 1.0f);
  float cy = 0.5f * ((ty + 1.0f) * (float)IMG_H - 1.0f);

  float s0 = fabsf(scaling_raw[2 * g + 0] + 0.5f);
  float s1 = fabsf(scaling_raw[2 * g + 1] + 0.5f);

  float theta = 6.28318530718f / (1.0f + __expf(-rotation_raw[g]));
  float co = __cosf(theta), si = __sinf(theta);

  float fx = __expf(normf_raw[2 * g + 0]);
  float fy = __expf(normf_raw[2 * g + 1]);

  float sx2 = s0 * s0, sy2 = s1 * s1;
  float a = co * co * sx2 + si * si * sy2;
  float b = co * si * (sx2 - sy2);
  float c = si * si * sx2 + co * co * sy2;
  float invd = 1.0f / (a * c - b * b);
  float conA = c * invd, conB = -b * invd, conC = a * invd;

  float op = opacity[g];
  float radius;
  if (op > ALPHA_THRESH) {
    // alpha > 1/255 <=> sigma < ln(255*op); max extent = sqrt(2*tau)*smax
    radius = sqrtf(2.0f * __logf(op * 255.0f)) * fmaxf(s0, s1) + 1e-3f;
  } else {
    radius = -1.0f;  // never contributes
  }

  gA[g] = make_float4(cx, cy, conA, conB);
  gB[g] = make_float4(conC, fx, fy, op);
  gC[g] = make_float4(features[3 * g + 0], features[3 * g + 1],
                      features[3 * g + 2], radius);
}

// ---------------------------------------------------------------------------
// One WMMA group: 32 survivors x 16 pixels (one wave). Builds the 16x32 f16
// A (wavelet weights) in contiguous k order (k = 16h + s); the 32x16 f16 B
// (colors in columns 0..2) is fetched pre-packed from component-major f16
// LDS (comp = min(col,3); comp 3 is stored zero) -- one aligned ds_load_b32
// per packed dword, fully branchless.
// ---------------------------------------------------------------------------
__device__ __forceinline__ v8f wipes_group(
    v8f acc, int g0, int half, int comp, float px, float py,
    const float4* __restrict__ sA, const float4* __restrict__ sB,
    const _Float16* __restrict__ sColH) {
  v16h a;
  union { v16h h; unsigned int u[8]; } bvu;
  // B element pair base for this lane's K-half (always even: g0 % 32 == 0)
  const unsigned int* colPairs =
      (const unsigned int*)(sColH + comp * CAP + g0 + 16 * half);
#pragma unroll
  for (int j = 0; j < 8; ++j) {
    // A layout (16-bit, 16x32): VGPR j -> K=2j,2j+1 (contiguous, matching B)
    // lanes 16-31 shifted by +16 in K.
    int kp = 2 * j + 16 * half;
#pragma unroll
    for (int e = 0; e < 2; ++e) {
      int k = g0 + kp + e;
      float4 A4 = sA[k];  // cx, cy, conA, conB
      float4 B4 = sB[k];  // conC, fx, fy, op
      float dx = px - A4.x;
      float dy = py - A4.y;
      float sigma = 0.5f * (A4.z * dx * dx + B4.x * dy * dy) + A4.w * dx * dy;
      float alpha = B4.w * __expf(-sigma);
      // Branchless: evaluate the wavelet term unconditionally, then cndmask.
      float cosv = __cosf(B4.y * dx + B4.z * dy);
      float wv = alpha * cosv;
      float w = (alpha > ALPHA_THRESH) ? wv : 0.0f;
      a[2 * j + e] = (_Float16)w;
    }
    // B layout (16-bit, 32x16): lanes 0-15 hold K=0..15, lanes 16-31 K=16..31,
    // two f16 K-elements per dword: one ds_load_b32 per packed pair.
    bvu.u[j] = colPairs[j];
  }
  return __builtin_amdgcn_wmma_f32_16x16x32_f16(
      /*neg_a=*/false, a, /*neg_b=*/false, bvu.h,
      /*c_mod=*/(short)0, acc, /*reuse_a=*/false, /*reuse_b=*/false);
}

// ---------------------------------------------------------------------------
// Render: one block = one 16x8 pixel tile, 8 waves, wave w owns image row
// y0+w (16 pixels). Streams gaussians in chunks of 256, compacts tile-hitting
// survivors into LDS (carry-over across chunks), runs full groups of 32
// through WMMA, flushes the remainder once at the end.
// ---------------------------------------------------------------------------
__global__ __launch_bounds__(256) void wipes_render(
    const float4* __restrict__ gA, const float4* __restrict__ gB,
    const float4* __restrict__ gC, float* __restrict__ out, int n) {
  __shared__ float4 sA[CAP];
  __shared__ float4 sB[CAP];
  __shared__ _Float16 sColH[4 * CAP] __attribute__((aligned(4)));
  __shared__ int sCount;

  const int tid = threadIdx.x;
  const int lane = tid & 31;
  const int wave = tid >> 5;
  const int half = lane >> 4;
  const int col = lane & 15;       // WMMA column / pixel row index M
  const int comp = (col < 3) ? col : 3;  // B color component (3 => zeros)

  const int x0 = blockIdx.x * TILE_W;
  const int y0 = blockIdx.y * TILE_H;
  const float px = (float)(x0 + col);
  const float py = (float)(y0 + wave);

  const float bx0 = (float)x0, bx1 = (float)(x0 + TILE_W - 1);
  const float by0 = (float)y0, by1 = (float)(y0 + TILE_H - 1);

  v8f acc = {};
  if (tid == 0) sCount = 0;
  __syncthreads();

  for (int base = 0; base < n; base += 256) {
    // --- cull: circle(center, radius) vs tile AABB of pixel centers ---
    int g = base + tid;  // n is a multiple of 256
    // Prefetch next chunk's cull stream (global_prefetch_b8); hidden behind
    // this chunk's WMMA group work.
    if (base + 256 < n) {
      __builtin_prefetch(&gA[g + 256], 0, 1);
      __builtin_prefetch(&gC[g + 256], 0, 1);
    }
    float4 A4 = gA[g];
    float4 C4 = gC[g];
    float r = C4.w;
    float qx = fminf(fmaxf(A4.x, bx0), bx1);
    float qy = fminf(fmaxf(A4.y, by0), by1);
    float ddx = A4.x - qx, ddy = A4.y - qy;
    bool hit = (r > 0.0f) && (ddx * ddx + ddy * ddy <= r * r);
    if (hit) {
      int slot = atomicAdd(&sCount, 1);
      sA[slot] = A4;
      sB[slot] = gB[g];
      sColH[0 * CAP + slot] = (_Float16)C4.x;
      sColH[1 * CAP + slot] = (_Float16)C4.y;
      sColH[2 * CAP + slot] = (_Float16)C4.z;
      sColH[3 * CAP + slot] = (_Float16)0.0f;
    }
    __syncthreads();

    int cnt = sCount;           // uniform across block
    int nGroups = cnt >> 5;
    for (int grp = 0; grp < nGroups; ++grp)
      acc = wipes_group(acc, grp * 32, half, comp, px, py, sA, sB, sColH);
    __syncthreads();            // all waves done reading before compaction

    int rem = cnt & 31;         // carry remainder to the front
    if (tid < rem) {
      int src = nGroups * 32 + tid;
      sA[tid] = sA[src];
      sB[tid] = sB[src];
      sColH[0 * CAP + tid] = sColH[0 * CAP + src];
      sColH[1 * CAP + tid] = sColH[1 * CAP + src];
      sColH[2 * CAP + tid] = sColH[2 * CAP + src];
    }
    if (tid == 0) sCount = rem;
    __syncthreads();
  }

  // --- flush remainder (pad with neutral gaussians: op=0 -> w=0) ---
  int cnt = sCount;
  if (cnt > 0) {
    if (tid >= cnt && tid < 32) {
      sA[tid] = make_float4(0.0f, 0.0f, 1.0f, 0.0f);
      sB[tid] = make_float4(1.0f, 0.0f, 0.0f, 0.0f);
      sColH[0 * CAP + tid] = (_Float16)0.0f;
      sColH[1 * CAP + tid] = (_Float16)0.0f;
      sColH[2 * CAP + tid] = (_Float16)0.0f;
      sColH[3 * CAP + tid] = (_Float16)0.0f;
    }
    __syncthreads();
    acc = wipes_group(acc, 0, half, comp, px, py, sA, sB, sColH);
  }

  // --- writeout: C layout -> lane c<3 holds channel c, VGPR j = row j+8*half
  // output shape (1,3,H,W): out[(ch*H + y)*W + x]
  if (col < 3) {
    int y = y0 + wave;
#pragma unroll
    for (int j = 0; j < 8; ++j) {
      int x = x0 + j + 8 * half;
      float v = fminf(fmaxf(acc[j], 0.0f), 1.0f);
      out[(col * IMG_H + y) * IMG_W + x] = v;
    }
  }
}

// ---------------------------------------------------------------------------
extern "C" void kernel_launch(void* const* d_in, const int* in_sizes, int n_in,
                              void* d_out, int out_size, void* d_ws,
                              size_t ws_size, hipStream_t stream) {
  const float* xyz_raw      = (const float*)d_in[0];  // (N,2)
  const float* scaling_raw  = (const float*)d_in[1];  // (N,2)
  const float* rotation_raw = (const float*)d_in[2];  // (N,1)
  const float* normf_raw    = (const float*)d_in[3];  // (N,2)
  const float* features     = (const float*)d_in[4];  // (N,3)
  const float* opacity      = (const float*)d_in[5];  // (N,1)

  const int n = in_sizes[0] / 2;  // N = 4096

  float4* gA = (float4*)d_ws;     // 3 * N * 16 bytes = 196608 B total
  float4* gB = gA + n;
  float4* gC = gB + n;

  wipes_preprocess<<<(n + 255) / 256, 256, 0, stream>>>(
      xyz_raw, scaling_raw, rotation_raw, normf_raw, features, opacity,
      gA, gB, gC, n);

  dim3 grid(IMG_W / TILE_W, IMG_H / TILE_H);  // 32 x 64 tiles
  wipes_render<<<grid, 256, 0, stream>>>(gA, gB, gC, (float*)d_out, n);
}
